// SelectiveSSMBlock_52536039964823
// MI455X (gfx1250) — matrix operand order repaired and checked by reading, not compiled, not run on hardware
//
#include <hip/hip_runtime.h>
#include <hip/hip_bf16.h>
#include <math.h>

// Problem constants (match reference)
#define BSZ 4
#define SEQ 1024
#define DM  256      // d_model == d_inner == d_state
#define DS2 512      // 2*D
#define KC  4        // conv kernel

typedef __attribute__((ext_vector_type(16))) _Float16 v16h;
typedef __attribute__((ext_vector_type(8)))  _Float16 v8h;
typedef __attribute__((ext_vector_type(8)))  float    v8f;

// ---------------------------------------------------------------------------
// WMMA fragment loaders (wave32, 16x16x32 f16).
// A-operand 16x32 (MxK): lane L holds row M = L&15; lanes 0-15 carry K chunks
// [k0..k0+7] and [k0+16..k0+23]; lanes 16-31 carry [k0+8..+15] and [k0+24..+31].
// B-operand 32x16 (KxN): lane L holds column N = L&15; same K chunks as A:
// lanes 0-15 carry [k0..k0+7] and [k0+16..k0+23]; lanes 16-31 carry [k0+8..+15] and [k0+24..+31].
// Weights are stored row-major (N x K), so B fragments are contiguous loads.
// ---------------------------------------------------------------------------
__device__ inline v16h load_a_frag(const _Float16* A, int lda, int m0, int k0) {
  int lane = threadIdx.x & 31;
  int m  = m0 + (lane & 15);
  int kb = k0 + ((lane >> 4) << 3);           // +0 or +8
  const _Float16* p = A + (size_t)m * lda + kb;
  v8h lo = *(const v8h*)p;
  v8h hi = *(const v8h*)(p + 16);
  v16h r;
#pragma unroll
  for (int i = 0; i < 8; ++i) { r[i] = lo[i]; r[i + 8] = hi[i]; }
  return r;
}

__device__ inline v16h load_b_frag(const _Float16* B, int ldb, int n0, int k0) {
  int lane = threadIdx.x & 31;
  int n  = n0 + (lane & 15);
  int kb = k0 + ((lane >> 4) << 3);           // +0 or +8
  const _Float16* p = B + (size_t)n * ldb + kb;
  v8h lo = *(const v8h*)p;
  v8h hi = *(const v8h*)(p + 16);
  v16h r;
#pragma unroll
  for (int i = 0; i < 8; ++i) { r[i] = lo[i]; r[i + 8] = hi[i]; }
  return r;
}

__device__ inline float silu_f(float v) { return v / (1.f + __expf(-v)); }

// ---------------------------------------------------------------------------
// Prep: convert weights to f16 (row-major N x K kept as-is), transpose A,
// and fold the C-projection into a single column-sum vector:
//   Csum_t = xc_t . colsum(W_C) + sum(b_C)
// ---------------------------------------------------------------------------
__global__ void prep_kernel(const float* __restrict__ W_in, const float* __restrict__ A,
                            const float* __restrict__ W_B, const float* __restrict__ W_C,
                            const float* __restrict__ W_out, const float* __restrict__ b_C,
                            _Float16* W_in_h, _Float16* A_t_h, _Float16* W_B_h,
                            _Float16* W_out_h, float* wcsum) {
  int stride = gridDim.x * blockDim.x;
  int t0 = blockIdx.x * blockDim.x + threadIdx.x;
  for (int i = t0; i < DS2 * DM; i += stride) W_in_h[i] = (_Float16)W_in[i];
  for (int i = t0; i < DM * DM; i += stride) {
    int j = i / DM, k = i % DM;                  // A[j][k] -> A_t[k][j]
    A_t_h[(size_t)k * DM + j] = (_Float16)A[i];
  }
  for (int i = t0; i < DM * DM; i += stride) W_B_h[i]  = (_Float16)W_B[i];
  for (int i = t0; i < DM * DM; i += stride) W_out_h[i] = (_Float16)W_out[i];
  for (int d = t0; d < DM; d += stride) {
    float s = 0.f;
    for (int k = 0; k < DM; ++k) s += W_C[(size_t)k * DM + d];
    wcsum[d] = s;
  }
  if (t0 == 0) {
    float s = 0.f;
    for (int k = 0; k < DM; ++k) s += b_C[k];
    wcsum[DM] = s;
  }
}

// ---------------------------------------------------------------------------
// LayerNorm: one wave per (b,s) row of 256, write f16 normalized activations.
// ---------------------------------------------------------------------------
__global__ __launch_bounds__(256) void ln_kernel(const float* __restrict__ x,
                                                 const float* __restrict__ g,
                                                 const float* __restrict__ b,
                                                 _Float16* __restrict__ xn_h) {
  int row  = (blockIdx.x * blockDim.x + threadIdx.x) >> 5;
  int lane = threadIdx.x & 31;
  const float* p = x + (size_t)row * DM;
  float vals[8], s = 0.f, s2 = 0.f;
#pragma unroll
  for (int i = 0; i < 8; ++i) {
    float v = p[lane + 32 * i];
    vals[i] = v; s += v; s2 += v * v;
  }
#pragma unroll
  for (int o = 16; o > 0; o >>= 1) { s += __shfl_xor(s, o, 32); s2 += __shfl_xor(s2, o, 32); }
  float mu  = s * (1.f / DM);
  float var = s2 * (1.f / DM) - mu * mu;
  float rs  = rsqrtf(var + 1e-5f);
#pragma unroll
  for (int i = 0; i < 8; ++i) {
    int d = lane + 32 * i;
    xn_h[(size_t)row * DM + d] = (_Float16)((vals[i] - mu) * rs * g[d] + b[d]);
  }
}

// ---------------------------------------------------------------------------
// GEMM1: xp = xn @ W_in^T + b_in  (4096 x 512, K=256).
// n < 256 -> xpr (f32, pre-conv); n >= 256 -> silu(gate) (f32).
// One 16x16 tile per wave, 8 waves per block.
// ---------------------------------------------------------------------------
__global__ __launch_bounds__(256) void gemm_in_kernel(const _Float16* __restrict__ Ah,
                                                      const _Float16* __restrict__ Bh,
                                                      const float* __restrict__ bias,
                                                      float* __restrict__ xpr,
                                                      float* __restrict__ sg) {
  int wid = (blockIdx.x * blockDim.x + threadIdx.x) >> 5;
  int n0 = (wid & 31) << 4;         // 32 N tiles
  int m0 = (wid >> 5) << 4;         // 256 M tiles
  v8f acc = {};
#pragma unroll
  for (int k0 = 0; k0 < DM; k0 += 32) {
    v16h a = load_a_frag(Ah, DM, m0, k0);
    v16h b = load_b_frag(Bh, DM, n0, k0);
    acc = __builtin_amdgcn_wmma_f32_16x16x32_f16(false, a, false, b, (short)0, acc, false, false);
  }
  int lane = threadIdx.x & 31;
  int n  = n0 + (lane & 15);
  int mb = (lane >> 4) << 3;
#pragma unroll
  for (int r = 0; r < 8; ++r) {
    int m = m0 + mb + r;
    float v = acc[r] + bias[n];
    if (n < DM) xpr[(size_t)m * DM + n] = v;
    else        sg[(size_t)m * DM + (n - DM)] = silu_f(v);
  }
}

// ---------------------------------------------------------------------------
// Causal depthwise conv (K=4) + SiLU -> xc (f32) and xc_h (f16).
// ---------------------------------------------------------------------------
__global__ void conv_kernel(const float* __restrict__ xpr, const float* __restrict__ cw,
                            const float* __restrict__ cb, float* __restrict__ xc,
                            _Float16* __restrict__ xc_h) {
  int idx = blockIdx.x * blockDim.x + threadIdx.x;      // over B*S*D
  if (idx >= BSZ * SEQ * DM) return;
  int d = idx & (DM - 1);
  int s = (idx >> 8) & (SEQ - 1);
  int base = idx - (s << 8) - d;                        // batch base
  float acc = cb[d];
#pragma unroll
  for (int k = 0; k < KC; ++k) {
    int sp = s - (KC - 1) + k;
    if (sp >= 0) acc += xpr[base + (sp << 8) + d] * cw[d * KC + k];
  }
  float v = silu_f(acc);
  xc[idx] = v;
  xc_h[idx] = (_Float16)v;
}

// ---------------------------------------------------------------------------
// Csum_t = xc_t . wcsum + bcsum  (one wave per row).
// ---------------------------------------------------------------------------
__global__ __launch_bounds__(256) void csum_kernel(const float* __restrict__ xc,
                                                   const float* __restrict__ wcsum,
                                                   float* __restrict__ Csum) {
  int row  = (blockIdx.x * blockDim.x + threadIdx.x) >> 5;
  int lane = threadIdx.x & 31;
  const float* p = xc + (size_t)row * DM;
  float s = 0.f;
#pragma unroll
  for (int i = 0; i < 8; ++i) s += p[lane + 32 * i] * wcsum[lane + 32 * i];
#pragma unroll
  for (int o = 16; o > 0; o >>= 1) s += __shfl_xor(s, o, 32);
  if (lane == 0) Csum[row] = s + wcsum[DM];
}

// ---------------------------------------------------------------------------
// GEMM (4096 x 256, K=256): Bm = xc @ W_B^T + b_B.
// ---------------------------------------------------------------------------
__global__ __launch_bounds__(256) void gemm_B_kernel(const _Float16* __restrict__ Ah,
                                                     const _Float16* __restrict__ Bh,
                                                     const float* __restrict__ bias,
                                                     float* __restrict__ Bm) {
  int wid = (blockIdx.x * blockDim.x + threadIdx.x) >> 5;
  int n0 = (wid & 15) << 4;
  int m0 = (wid >> 4) << 4;
  v8f acc = {};
#pragma unroll
  for (int k0 = 0; k0 < DM; k0 += 32) {
    v16h a = load_a_frag(Ah, DM, m0, k0);
    v16h b = load_b_frag(Bh, DM, n0, k0);
    acc = __builtin_amdgcn_wmma_f32_16x16x32_f16(false, a, false, b, (short)0, acc, false, false);
  }
  int lane = threadIdx.x & 31;
  int n  = n0 + (lane & 15);
  int mb = (lane >> 4) << 3;
#pragma unroll
  for (int r = 0; r < 8; ++r) {
    int m = m0 + mb + r;
    Bm[(size_t)m * DM + n] = acc[r] + bias[n];
  }
}

// ---------------------------------------------------------------------------
// Sequential scan. Key identity: all D rows of h are equal, so the state is
// g_t (B x DS) with g_t = g_{t-1} @ A + B_t, and
//   y_t = Csum_t * g_t + D_p * xc_t, then y *= silu(gate).
// One workgroup, 16 waves; batches 0-3 occupy rows 0-3 of a 16-row WMMA tile
// (rows 4-15 stay zero). Each wave owns 16 state columns; the constant A^T
// fragments live in registers; f16 state ping-pongs in LDS (1 barrier/step).
// Latency tuning: two independent 4-deep WMMA chains (halves the RAW hazard
// depth) and per-step global operands hoisted above the WMMAs so HBM/L2
// latency overlaps the matrix ops.
// ---------------------------------------------------------------------------
__global__ __launch_bounds__(512) void scan_kernel(const _Float16* __restrict__ At_h,
                                                   const float* __restrict__ Bm,
                                                   const float* __restrict__ Csum,
                                                   const float* __restrict__ xc,
                                                   const float* __restrict__ Dp,
                                                   const float* __restrict__ sg,
                                                   _Float16* __restrict__ y_h) {
  __shared__ _Float16 G[2][16][DM];                    // 16 KB ping-pong state
  int tid  = threadIdx.x;
  int wave = tid >> 5;
  int lane = tid & 31;
  int n0 = wave << 4;                                  // 16 columns per wave

  v16h bfrag[8];
#pragma unroll
  for (int kk = 0; kk < 8; ++kk) bfrag[kk] = load_b_frag(At_h, DM, n0, kk * 32);

  for (int i = tid; i < 16 * DM; i += 512) (&G[0][0][0])[i] = (_Float16)0.f;
  __syncthreads();

  int n  = n0 + (lane & 15);
  int mb = (lane >> 4) << 3;                           // lanes 0-15: rows 0-7
  bool active = (mb == 0);                             // real batches live in r<4
  float dpn = Dp[n];
  int p = 0;
  for (int t = 0; t < SEQ; ++t) {
    // Hoisted per-step operands: issue global loads before the WMMA chain.
    float bv[4] = {0.f, 0.f, 0.f, 0.f};
    float xv[4] = {0.f, 0.f, 0.f, 0.f};
    float gv[4] = {0.f, 0.f, 0.f, 0.f};
    float cs[4] = {0.f, 0.f, 0.f, 0.f};
#pragma unroll
    for (int r = 0; r < BSZ; ++r) {
      if (active) {
        size_t idx = ((size_t)r * SEQ + t) * DM + n;
        bv[r] = Bm[idx];
        xv[r] = xc[idx];
        gv[r] = sg[idx];
        cs[r] = Csum[r * SEQ + t];
      }
    }
    // Two independent accumulation chains of 4 WMMAs each.
    v8f acc0 = {}, acc1 = {};
#pragma unroll
    for (int kk = 0; kk < 4; ++kk) {
      v16h a0 = load_a_frag(&G[p][0][0], DM, 0, kk * 32);
      v16h a1 = load_a_frag(&G[p][0][0], DM, 0, (kk + 4) * 32);
      acc0 = __builtin_amdgcn_wmma_f32_16x16x32_f16(false, a0, false, bfrag[kk], (short)0,
                                                    acc0, false, false);
      acc1 = __builtin_amdgcn_wmma_f32_16x16x32_f16(false, a1, false, bfrag[kk + 4], (short)0,
                                                    acc1, false, false);
    }
#pragma unroll
    for (int r = 0; r < 8; ++r) {
      int m = mb + r;
      float v = acc0[r] + acc1[r];
      if (active && r < BSZ) {
        v += bv[r];                                    // g_t = g_{t-1} A + B_t
        size_t idx = ((size_t)r * SEQ + t) * DM + n;
        float y = cs[r] * v + dpn * xv[r];
        y_h[idx] = (_Float16)(y * gv[r]);              // gated output, f16
      }
      G[1 - p][m][n] = (_Float16)v;                    // rows >=4 remain 0
    }
    __syncthreads();
    p ^= 1;
  }
}

// ---------------------------------------------------------------------------
// GEMM out (4096 x 256, K=256): out = y @ W_out^T + b_out + residual(x).
// ---------------------------------------------------------------------------
__global__ __launch_bounds__(256) void gemm_out_kernel(const _Float16* __restrict__ Ah,
                                                       const _Float16* __restrict__ Bh,
                                                       const float* __restrict__ bias,
                                                       const float* __restrict__ res,
                                                       float* __restrict__ out) {
  int wid = (blockIdx.x * blockDim.x + threadIdx.x) >> 5;
  int n0 = (wid & 15) << 4;
  int m0 = (wid >> 4) << 4;
  v8f acc = {};
#pragma unroll
  for (int k0 = 0; k0 < DM; k0 += 32) {
    v16h a = load_a_frag(Ah, DM, m0, k0);
    v16h b = load_b_frag(Bh, DM, n0, k0);
    acc = __builtin_amdgcn_wmma_f32_16x16x32_f16(false, a, false, b, (short)0, acc, false, false);
  }
  int lane = threadIdx.x & 31;
  int n  = n0 + (lane & 15);
  int mb = (lane >> 4) << 3;
#pragma unroll
  for (int r = 0; r < 8; ++r) {
    int m = m0 + mb + r;
    out[(size_t)m * DM + n] = acc[r] + bias[n] + res[(size_t)m * DM + n];
  }
}

// ---------------------------------------------------------------------------
// Launch
// ---------------------------------------------------------------------------
static inline size_t alignup(size_t v) { return (v + 255) & ~(size_t)255; }

extern "C" void kernel_launch(void* const* d_in, const int* in_sizes, int n_in,
                              void* d_out, int out_size, void* d_ws, size_t ws_size,
                              hipStream_t stream) {
  const float* x      = (const float*)d_in[0];
  const float* ln_g   = (const float*)d_in[1];
  const float* ln_b   = (const float*)d_in[2];
  const float* W_in   = (const float*)d_in[3];
  const float* b_in   = (const float*)d_in[4];
  const float* conv_w = (const float*)d_in[5];
  const float* conv_b = (const float*)d_in[6];
  const float* A      = (const float*)d_in[7];
  const float* W_B    = (const float*)d_in[8];
  const float* b_B    = (const float*)d_in[9];
  const float* W_C    = (const float*)d_in[10];
  const float* b_C    = (const float*)d_in[11];
  const float* D_p    = (const float*)d_in[12];
  const float* W_out  = (const float*)d_in[13];
  const float* b_out  = (const float*)d_in[14];
  (void)in_sizes; (void)n_in; (void)out_size; (void)ws_size;

  const size_t ROWS = (size_t)BSZ * SEQ;               // 4096
  char* ws = (char*)d_ws;
  size_t off = 0;
  _Float16* W_in_h  = (_Float16*)(ws + off); off = alignup(off + DS2 * DM * sizeof(_Float16));
  _Float16* A_t_h   = (_Float16*)(ws + off); off = alignup(off + DM * DM * sizeof(_Float16));
  _Float16* W_B_h   = (_Float16*)(ws + off); off = alignup(off + DM * DM * sizeof(_Float16));
  _Float16* W_out_h = (_Float16*)(ws + off); off = alignup(off + DM * DM * sizeof(_Float16));
  float*    wcsum   = (float*)   (ws + off); off = alignup(off + (DM + 1) * sizeof(float));
  _Float16* xn_h    = (_Float16*)(ws + off); off = alignup(off + ROWS * DM * sizeof(_Float16));
  float*    xpr     = (float*)   (ws + off); off = alignup(off + ROWS * DM * sizeof(float));
  float*    sg      = (float*)   (ws + off); off = alignup(off + ROWS * DM * sizeof(float));
  float*    xc      = (float*)   (ws + off); off = alignup(off + ROWS * DM * sizeof(float));
  _Float16* xc_h    = (_Float16*)(ws + off); off = alignup(off + ROWS * DM * sizeof(_Float16));
  float*    Bm      = (float*)   (ws + off); off = alignup(off + ROWS * DM * sizeof(float));
  float*    Csum    = (float*)   (ws + off); off = alignup(off + ROWS * sizeof(float));
  _Float16* y_h     = (_Float16*)(ws + off); off = alignup(off + ROWS * DM * sizeof(_Float16));

  prep_kernel<<<64, 256, 0, stream>>>(W_in, A, W_B, W_C, W_out, b_C,
                                      W_in_h, A_t_h, W_B_h, W_out_h, wcsum);
  ln_kernel<<<512, 256, 0, stream>>>(x, ln_g, ln_b, xn_h);
  gemm_in_kernel<<<1024, 256, 0, stream>>>(xn_h, W_in_h, b_in, xpr, sg);
  conv_kernel<<<(BSZ * SEQ * DM + 255) / 256, 256, 0, stream>>>(xpr, conv_w, conv_b, xc, xc_h);
  csum_kernel<<<512, 256, 0, stream>>>(xc, wcsum, Csum);
  gemm_B_kernel<<<512, 256, 0, stream>>>(xc_h, W_B_h, b_B, Bm);
  scan_kernel<<<1, 512, 0, stream>>>(A_t_h, Bm, Csum, xc, D_p, sg, y_h);
  gemm_out_kernel<<<512, 256, 0, stream>>>(y_h, W_out_h, b_out, x, (float*)d_out);
}
